// DotProductAttention_64149631533643
// MI455X (gfx1250) — matrix operand order repaired and checked by reading, not compiled, not run on hardware
//
#include <hip/hip_runtime.h>
#include <hip/hip_bf16.h>

typedef __attribute__((ext_vector_type(16))) _Float16 v16h;
typedef __attribute__((ext_vector_type(8)))  _Float16 v8h;
typedef __attribute__((ext_vector_type(8)))  float    v8f;

#define DIM    64      // head dim
#define KT     64      // keys per softmax step (2 WMMA K-chunks of 32)
#define QW     16      // queries per wave (WMMA M/N)
#define WAVES  8
#define BLOCKT (WAVES * 32)
#define QB     (QW * WAVES)   // 128 queries per block
#define NEG_BIG (-1e6f)
#define LOG2E  1.4426950408889634f

union V16 { v16h v; v8h h[2]; };

__global__ __launch_bounds__(BLOCKT)
void flash_attn_f16wmma(const float* __restrict__ Qg,
                        const float* __restrict__ Kg,
                        const float* __restrict__ Vg,
                        const int*   __restrict__ VL,
                        float* __restrict__ Og,
                        int seq)   // 2048
{
    // LDS: K row-major f16 [key][dim] 8KB, V^T f16 [dim][key] 8KB, P scratch 16KB
    __shared__ __align__(16) _Float16 Ksh[KT * DIM];
    __shared__ __align__(16) _Float16 Vsh[DIM * KT];
    __shared__ __align__(16) _Float16 Psh[WAVES][QW * KT];

    const int tid  = threadIdx.x;
    const int wave = tid >> 5;
    const int lane = tid & 31;
    const int ln   = lane & 15;      // column / M index inside 16
    const int g    = lane >> 4;      // lane-group (selects k/r offset)

    const int batch = blockIdx.x;
    const int qbase = blockIdx.y * QB + wave * QW;

    const int vlen = VL[batch];
    const int jmax = (vlen + KT - 1) / KT;     // key tiles actually needed

    const size_t bOff = (size_t)batch * seq * DIM;

    // ---- Q fragments (A layout), scaled by log2(e)/sqrt(64), f16 ----
    // A layout: lane m = ln; half e -> k = (e/8)*16 + g*8 + (e%8)
    V16 qf[2];
    {
        const float qs = 0.125f * LOG2E;       // base-2 softmax domain
        const float* qrow = Qg + bOff + (size_t)(qbase + ln) * DIM;
        #pragma unroll
        for (int c = 0; c < 2; ++c) {
            const float* b0 = qrow + c * 32 + g * 8;
            const float* b1 = qrow + c * 32 + 16 + g * 8;
            #pragma unroll
            for (int i = 0; i < 8; ++i) {
                qf[c].v[i]     = (_Float16)(b0[i] * qs);
                qf[c].v[8 + i] = (_Float16)(b1[i] * qs);
            }
        }
    }

    // ---- flash state ----
    v8f Ofr[4];
    #pragma unroll
    for (int t = 0; t < 4; ++t) Ofr[t] = (v8f){};
    float mrow[8], lrow[8];
    #pragma unroll
    for (int v = 0; v < 8; ++v) { mrow[v] = -INFINITY; lrow[v] = 0.0f; }

    // ---- cooperative loader: each thread owns one key row segment ----
    const int key = tid >> 2;            // 0..63
    const int ds  = (tid & 3) * 16;      // dim segment {0,16,32,48}
    float kreg[16], vreg[16];

    auto fetch_tile = [&](int kbase) {
        const float* kp = Kg + bOff + (size_t)(kbase + key) * DIM + ds;
        const float* vp = Vg + bOff + (size_t)(kbase + key) * DIM + ds;
        #pragma unroll
        for (int i = 0; i < 16; ++i) { kreg[i] = kp[i]; vreg[i] = vp[i]; }
    };

    fetch_tile(0);   // prefetch tile 0 (vlen >= 1 guarantees jmax >= 1)

    for (int j = 0; j < jmax; ++j) {
        const int kbase = j * KT;
        __syncthreads();   // previous tile's consumers are done with LDS

        // ---- stage prefetched registers into LDS (f32 -> f16) ----
        {
            V16 kh;
            #pragma unroll
            for (int i = 0; i < 16; ++i) kh.v[i] = (_Float16)kreg[i];
            *(v8h*)&Ksh[key * DIM + ds]     = kh.h[0];
            *(v8h*)&Ksh[key * DIM + ds + 8] = kh.h[1];
            #pragma unroll
            for (int i = 0; i < 16; ++i)              // transposed scatter
                Vsh[(ds + i) * KT + key] = (_Float16)vreg[i];
        }
        // ---- overlap: issue next tile's global loads during compute ----
        if (j + 1 < jmax) fetch_tile((j + 1) * KT);
        __syncthreads();

        // ---- S = Q * K^T : four 16x16 f32 tiles (key quarters) ----
        // B layout: lane n = ln (key), half e -> r = (e/8)*16 + g*8 + (e%8) (dim within chunk)
        v8f S[4];
        #pragma unroll
        for (int t = 0; t < 4; ++t) {
            v8f acc = (v8f){};
            #pragma unroll
            for (int c = 0; c < 2; ++c) {
                V16 bf;
                const int kk = t * 16 + ln;
                bf.h[0] = *(const v8h*)&Ksh[kk * DIM + c * 32 + g * 8];
                bf.h[1] = *(const v8h*)&Ksh[kk * DIM + c * 32 + 16 + g * 8];
                acc = __builtin_amdgcn_wmma_f32_16x16x32_f16(
                        false, qf[c].v, false, bf.v, (short)0, acc, false, false);
            }
            S[t] = acc;
        }

        // ---- boundary masking (block-uniform branch) ----
        if (kbase + KT > vlen) {
            #pragma unroll
            for (int t = 0; t < 4; ++t) {
                const bool ok = (kbase + t * 16 + ln) < vlen;
                #pragma unroll
                for (int v = 0; v < 8; ++v)
                    S[t][v] = ok ? S[t][v] : NEG_BIG;
            }
        }

        // ---- online softmax (base-2): row max over 64 cols ----
        float alpha[8];
        #pragma unroll
        for (int v = 0; v < 8; ++v) {
            float mx = fmaxf(fmaxf(S[0][v], S[1][v]), fmaxf(S[2][v], S[3][v]));
            mx = fmaxf(mx, __shfl_xor(mx, 1, 32));
            mx = fmaxf(mx, __shfl_xor(mx, 2, 32));
            mx = fmaxf(mx, __shfl_xor(mx, 4, 32));
            mx = fmaxf(mx, __shfl_xor(mx, 8, 32));
            const float mnew = fmaxf(mrow[v], mx);
            alpha[v] = __builtin_amdgcn_exp2f(mrow[v] - mnew);
            mrow[v]  = mnew;
        }

        // P = exp2(S - m); update per-lane partial row sums
        #pragma unroll
        for (int t = 0; t < 4; ++t)
            #pragma unroll
            for (int v = 0; v < 8; ++v)
                S[t][v] = __builtin_amdgcn_exp2f(S[t][v] - mrow[v]);
        #pragma unroll
        for (int v = 0; v < 8; ++v)
            lrow[v] = lrow[v] * alpha[v] + ((S[0][v] + S[1][v]) + (S[2][v] + S[3][v]));

        // ---- transpose P (C layout) -> A layout via per-wave LDS scratch ----
        // C layout element (m = v + 8g, col = t*16 + ln)
        #pragma unroll
        for (int t = 0; t < 4; ++t)
            #pragma unroll
            for (int v = 0; v < 8; ++v)
                Psh[wave][(v + 8 * g) * KT + t * 16 + ln] = (_Float16)S[t][v];
        // same-wave DS ops are in-order: no barrier needed
        V16 pf[2];
        #pragma unroll
        for (int h = 0; h < 2; ++h) {
            pf[h].h[0] = *(const v8h*)&Psh[wave][ln * KT + h * 32 + g * 8];
            pf[h].h[1] = *(const v8h*)&Psh[wave][ln * KT + h * 32 + 16 + g * 8];
        }

        // ---- O = O*alpha + P*V  (two 32-key chunks) ----
        #pragma unroll
        for (int t = 0; t < 4; ++t)
            #pragma unroll
            for (int v = 0; v < 8; ++v)
                Ofr[t][v] *= alpha[v];

        // B = V: lane n = dim col, half e -> r = key-in-chunk = (e/8)*16 + g*8 + (e%8)
        #pragma unroll
        for (int t = 0; t < 4; ++t) {
            const int dim = t * 16 + ln;
            #pragma unroll
            for (int h = 0; h < 2; ++h) {
                V16 vf;
                vf.h[0] = *(const v8h*)&Vsh[dim * KT + h * 32 + g * 8];
                vf.h[1] = *(const v8h*)&Vsh[dim * KT + h * 32 + 16 + g * 8];
                Ofr[t] = __builtin_amdgcn_wmma_f32_16x16x32_f16(
                           false, pf[h].v, false, vf.v, (short)0, Ofr[t], false, false);
            }
        }
    }

    // ---- final normalization: reduce partial row sums across 16 lanes ----
    float inv[8];
    #pragma unroll
    for (int v = 0; v < 8; ++v) {
        float s = lrow[v];
        s += __shfl_xor(s, 1, 32);
        s += __shfl_xor(s, 2, 32);
        s += __shfl_xor(s, 4, 32);
        s += __shfl_xor(s, 8, 32);
        inv[v] = 1.0f / s;
    }

    #pragma unroll
    for (int t = 0; t < 4; ++t)
        #pragma unroll
        for (int v = 0; v < 8; ++v) {
            const int m = v + 8 * g;
            Og[bOff + (size_t)(qbase + m) * DIM + t * 16 + ln] = Ofr[t][v] * inv[v];
        }
}

extern "C" void kernel_launch(void* const* d_in, const int* in_sizes, int n_in,
                              void* d_out, int out_size, void* d_ws, size_t ws_size,
                              hipStream_t stream) {
    (void)d_ws; (void)ws_size; (void)n_in; (void)out_size;
    const float* Q  = (const float*)d_in[0];
    const float* K  = (const float*)d_in[1];
    const float* V  = (const float*)d_in[2];
    const int*   VL = (const int*)d_in[3];
    float*       O  = (float*)d_out;

    const int n   = in_sizes[3];                 // 32 batches
    const int seq = in_sizes[0] / (n * DIM);     // 2048

    dim3 grid(n, seq / QB);                      // 32 x 16 blocks
    dim3 block(BLOCKT);                          // 256 threads = 8 waves
    flash_attn_f16wmma<<<grid, block, 0, stream>>>(Q, K, V, VL, O, seq);
}
